// RBFKanLayer_42502996361683
// MI455X (gfx1250) — matrix operand order repaired and checked by reading, not compiled, not run on hardware
//
#include <hip/hip_runtime.h>

// ---- CDNA5 / gfx1250 WMMA vector types ----
typedef __attribute__((ext_vector_type(16))) __bf16 v16bf;
typedef __attribute__((ext_vector_type(2)))  __bf16 bf16x2;
typedef __attribute__((ext_vector_type(8)))  float  v8f;

#define IN_F   1024
#define OUT_F  1024
#define GRID_N 20
#define BATCH  4096

#define BM 128     // block rows (batch): 8 waves x 16 rows
#define BN 128     // block cols (out features): 8 x 16 per wave

union BFrag { v16bf v; unsigned int w[8]; uint4 q[2]; };

__device__ __forceinline__ unsigned int pack_bf16(float a, float b) {
#if __has_builtin(__builtin_amdgcn_cvt_pk_bf16_f32)
    union { bf16x2 h2; unsigned int u; } p;
    p.h2 = __builtin_amdgcn_cvt_pk_bf16_f32(a, b);
    return p.u;
#else
    // force the fused packed convert (one instruction); the backend fails to
    // merge two independent fptrunc-to-bf16 into the pk form on its own
    unsigned int u;
    asm("v_cvt_pk_bf16_f32 %0, %1, %2" : "=v"(u) : "v"(a), "v"(b));
    return u;
#endif
}

// fast exp2 -> v_exp_f32 (negate folds into source modifier)
#if __has_builtin(__builtin_amdgcn_exp2f)
#define FAST_EXP2(x) __builtin_amdgcn_exp2f(x)
#else
#define FAST_EXP2(x) __expf((x) * 0.69314718056f)   // exp(x*ln2) == exp2(x)
#endif

__global__ __launch_bounds__(256)
void rbf_kan_wmma_kernel(const float* __restrict__ x,
                         const float* __restrict__ spline_w,   // [IN_F*GRID_N, OUT_F]
                         const float* __restrict__ base_w,     // [OUT_F, IN_F]
                         const float* __restrict__ grid,       // [GRID_N]
                         float* __restrict__ out)              // [BATCH, OUT_F]
{
    // x tile: 128 rows x 32 features (stride 36 -> conflict-free row reads)
    __shared__ float xs[BM][36];
    // weight tile, packed bf16 pairs along K: [n][kpair], stride 20 words
    // (20*row + 8*half) % 4 == 0 -> 16B-aligned b128 fragment reads
    __shared__ unsigned int wsp[2][BN][20];

    const int tid  = threadIdx.x;
    const int lane = tid & 31;
    const int wave = tid >> 5;      // 0..7 -> M strip wave*16
    const int half = lane >> 4;     // 0..1
    const int l16  = lane & 15;

    const int nBase = blockIdx.x * BN;
    const int mBase = blockIdx.y * BM;
    const int aRow  = wave * 16 + l16;

    // staging thread mappings
    const int sx_r  = tid >> 3;            // rows (+32 per t), cols sx_c4
    const int sx_c4 = (tid & 7) * 4;
    const int sp_p  = tid >> 4;            // spline: k-pair index 0..15
    const int sp_n4 = (tid & 15) * 4;      // spline: 4 consecutive n (+64 per t)

    // exp(-20 d^2) = exp2(-(s*x - s*g)^2), s = sqrt(20*log2(e))  (folded scale)
    const float sW = __builtin_sqrtf(20.0f * 1.44269504088896f);

    v8f acc[8] = {{}, {}, {}, {}, {}, {}, {}, {}};

    for (int i0 = 0; i0 < IN_F; i0 += 32) {
        // ---- stage x tile [128][32] (b128 coalesced) ----
#pragma unroll
        for (int t = 0; t < 4; ++t) {
            const int r = sx_r + t * 32;
            const float4 f4 = *(const float4*)(x + (size_t)(mBase + r) * IN_F + i0 + sx_c4);
            *(float4*)&xs[r][sx_c4] = f4;
        }
        // ---- stage base_w k-chunk into wsp[0], packed bf16 pairs along k ----
#pragma unroll
        for (int t = 0; t < 4; ++t) {
            const int n = sx_r + t * 32;   // 0..127
            const float4 f4 = *(const float4*)(base_w + (size_t)(nBase + n) * IN_F + i0 + sx_c4);
            uint2 pp;
            pp.x = pack_bf16(f4.x, f4.y);
            pp.y = pack_bf16(f4.z, f4.w);
            *(uint2*)&wsp[0][n][sx_c4 >> 1] = pp;
        }
        __syncthreads();

        // ---- pull this lane's 16 x-values into registers (reused 21 steps) ----
        float xv[16], xsv[16];
#pragma unroll
        for (int v = 0; v < 8; ++v) {
            const int kA = 2 * v + 16 * half;
            const float2 ap = *(const float2*)&xs[aRow][kA];
            xv[2 * v]      = ap.x;
            xv[2 * v + 1]  = ap.y;
            xsv[2 * v]     = ap.x * sW;    // pre-scaled for RBF steps
            xsv[2 * v + 1] = ap.y * sW;
        }

        // ---- issue loads for spline g=0 tile (hide behind base compute) ----
        float4 r0a, r1a, r0b, r1b;
        {
            const size_t row0 = (size_t)(i0 + 2 * sp_p) * GRID_N + 0;
            r0a = *(const float4*)(spline_w + row0 * OUT_F + nBase + sp_n4);
            r1a = *(const float4*)(spline_w + (row0 + GRID_N) * OUT_F + nBase + sp_n4);
            r0b = *(const float4*)(spline_w + row0 * OUT_F + nBase + sp_n4 + 64);
            r1b = *(const float4*)(spline_w + (row0 + GRID_N) * OUT_F + nBase + sp_n4 + 64);
        }

        // ---- step 0: base GEMM (A = bf16(x), B = base_w^T chunk) ----
        {
            v16bf a;
#pragma unroll
            for (int v = 0; v < 8; ++v) {
                a[2 * v]     = (__bf16)xv[2 * v];
                a[2 * v + 1] = (__bf16)xv[2 * v + 1];
            }
#pragma unroll
            for (int j = 0; j < 8; ++j) {
                BFrag B;
                const unsigned int* wrow = &wsp[0][j * 16 + l16][8 * half];
                B.q[0] = *(const uint4*)(wrow + 0);
                B.q[1] = *(const uint4*)(wrow + 4);
                acc[j] = __builtin_amdgcn_wmma_f32_16x16x32_bf16(false, a, false, B.v,
                                                                 (short)0, acc[j], false, false);
            }
        }
        // store g=0 tile into wsp[1]
        wsp[1][sp_n4 + 0][sp_p]  = pack_bf16(r0a.x, r1a.x);
        wsp[1][sp_n4 + 1][sp_p]  = pack_bf16(r0a.y, r1a.y);
        wsp[1][sp_n4 + 2][sp_p]  = pack_bf16(r0a.z, r1a.z);
        wsp[1][sp_n4 + 3][sp_p]  = pack_bf16(r0a.w, r1a.w);
        wsp[1][sp_n4 + 64][sp_p] = pack_bf16(r0b.x, r1b.x);
        wsp[1][sp_n4 + 65][sp_p] = pack_bf16(r0b.y, r1b.y);
        wsp[1][sp_n4 + 66][sp_p] = pack_bf16(r0b.z, r1b.z);
        wsp[1][sp_n4 + 67][sp_p] = pack_bf16(r0b.w, r1b.w);
        __syncthreads();

        // ---- 20 spline steps: fixed grid knot g, K = 32 features ----
#pragma unroll 1
        for (int g = 0; g < GRID_N; ++g) {
            const int buf = (g + 1) & 1;

            // issue next tile's loads early (g+1), prefetch g+2
            if (g < GRID_N - 1) {
                const size_t row0 = (size_t)(i0 + 2 * sp_p) * GRID_N + (g + 1);
                r0a = *(const float4*)(spline_w + row0 * OUT_F + nBase + sp_n4);
                r1a = *(const float4*)(spline_w + (row0 + GRID_N) * OUT_F + nBase + sp_n4);
                r0b = *(const float4*)(spline_w + row0 * OUT_F + nBase + sp_n4 + 64);
                r1b = *(const float4*)(spline_w + (row0 + GRID_N) * OUT_F + nBase + sp_n4 + 64);
            }
            if (g < GRID_N - 2) {
                const size_t row0 = (size_t)(i0 + 2 * sp_p) * GRID_N + (g + 2);
                __builtin_prefetch(spline_w + row0 * OUT_F + nBase + sp_n4, 0, 1);
            }

            const float sg = grid[g] * sW;   // uniform scalar load + scale

            // A fragment: RBF basis for 32 features at knot g
            // per element: v_sub, v_mul, v_exp (neg via src modifier), + pack
            v16bf a;
#pragma unroll
            for (int v = 0; v < 8; ++v) {
                const float d0 = xsv[2 * v]     - sg;
                const float d1 = xsv[2 * v + 1] - sg;
                a[2 * v]     = (__bf16)FAST_EXP2(-(d0 * d0));
                a[2 * v + 1] = (__bf16)FAST_EXP2(-(d1 * d1));
            }

            // 8 WMMAs sharing one A fragment
#pragma unroll
            for (int j = 0; j < 8; ++j) {
                BFrag B;
                const unsigned int* wrow = &wsp[buf][j * 16 + l16][8 * half];
                B.q[0] = *(const uint4*)(wrow + 0);
                B.q[1] = *(const uint4*)(wrow + 4);
                acc[j] = __builtin_amdgcn_wmma_f32_16x16x32_bf16(false, a, false, B.v,
                                                                 (short)0, acc[j], false, false);
            }

            // stash next tile
            if (g < GRID_N - 1) {
                wsp[buf ^ 1][sp_n4 + 0][sp_p]  = pack_bf16(r0a.x, r1a.x);
                wsp[buf ^ 1][sp_n4 + 1][sp_p]  = pack_bf16(r0a.y, r1a.y);
                wsp[buf ^ 1][sp_n4 + 2][sp_p]  = pack_bf16(r0a.z, r1a.z);
                wsp[buf ^ 1][sp_n4 + 3][sp_p]  = pack_bf16(r0a.w, r1a.w);
                wsp[buf ^ 1][sp_n4 + 64][sp_p] = pack_bf16(r0b.x, r1b.x);
                wsp[buf ^ 1][sp_n4 + 65][sp_p] = pack_bf16(r0b.y, r1b.y);
                wsp[buf ^ 1][sp_n4 + 66][sp_p] = pack_bf16(r0b.z, r1b.z);
                wsp[buf ^ 1][sp_n4 + 67][sp_p] = pack_bf16(r0b.w, r1b.w);
            }
            __syncthreads();
        }
    }

    // ---- epilogue: C/D layout (lane col = l16, VGPR v -> row v + 8*half) ----
    // streaming (non-temporal) stores: keep L2 for spline weights
#pragma unroll
    for (int j = 0; j < 8; ++j) {
        const int col = nBase + j * 16 + l16;
#pragma unroll
        for (int v = 0; v < 8; ++v) {
            const int row = mBase + wave * 16 + v + 8 * half;
            __builtin_nontemporal_store(acc[j][v], out + (size_t)row * OUT_F + col);
        }
    }
}

extern "C" void kernel_launch(void* const* d_in, const int* in_sizes, int n_in,
                              void* d_out, int out_size, void* d_ws, size_t ws_size,
                              hipStream_t stream)
{
    const float* x        = (const float*)d_in[0];
    const float* spline_w = (const float*)d_in[1];
    const float* base_w   = (const float*)d_in[2];
    const float* grid     = (const float*)d_in[3];
    float* out            = (float*)d_out;

    dim3 block(256);
    dim3 g(OUT_F / BN, BATCH / BM);   // 8 x 32 blocks
    rbf_kan_wmma_kernel<<<g, block, 0, stream>>>(x, spline_w, base_w, grid, out);
}
